// ScaledDotProductAttention_90761248899177
// MI455X (gfx1250) — matrix operand order repaired and checked by reading, not compiled, not run on hardware
//
#include <hip/hip_runtime.h>

typedef __attribute__((ext_vector_type(16))) _Float16 v16h;
typedef __attribute__((ext_vector_type(8)))  _Float16 v8h;
typedef __attribute__((ext_vector_type(8)))  float    v8f;

#define S_LEN 2048
#define D_LEN 64
#define NHEAD 32            // B*H = 2*16
#define WAVES 8             // 256 threads, wave32
#define BLOCKS_PER_HEAD (S_LEN / (16 * WAVES))   // 16

#define KPAD 72             // padded row stride (halves) for 64-wide tiles
#define VPAD 40             // padded row stride (halves) for 32-wide tiles

static __device__ __forceinline__ v16h cat8(v8h a, v8h b) {
    return __builtin_shufflevector(a, b, 0,1,2,3,4,5,6,7,8,9,10,11,12,13,14,15);
}
static __device__ __forceinline__ v8h ld8(const _Float16* p) {
    return *(const v8h*)p;   // 16B-aligned LDS load -> ds_load_b128
}

// -------- Pass 1: per-key-column max and exp-sum over all queries ----------
// scores[q,k] = (Q[q,:] . K[k,:]) / sf ; softmax is over q (columns of [q,k])
__global__ __launch_bounds__(256) void
attn_colstats_kernel(const float* __restrict__ Q, const float* __restrict__ K,
                     const float* __restrict__ sf_p,
                     float* __restrict__ m_ws, float* __restrict__ l_ws)
{
    __shared__ _Float16 sQ[16 * KPAD];   // shared scaled-Q f16 tile, row-major

    const int tid  = threadIdx.x;
    const int lane = tid & 31;
    const int wave = tid >> 5;
    const int head = blockIdx.x / BLOCKS_PER_HEAD;
    const int kb0  = (blockIdx.x % BLOCKS_PER_HEAD) * (16 * WAVES) + wave * 16;

    const float inv_sf = 1.0f / sf_p[0];
    const size_t hb = (size_t)head * S_LEN * D_LEN;
    const float* __restrict__ Qh = Q + hb;
    const float* __restrict__ Kh = K + hb;

    const int nl = lane & 15;
    const int hi = (lane >> 4) & 1;

    // B operand = K^T tile (this wave's 16 key columns), loaded once from global.
    // B k order used here (contiguous, same as A): lane&15 = n, VGPR r holds K = hi*16 + 2r, 2r+1
    v16h bk0, bk1;
    {
        const float* kp = Kh + (size_t)(kb0 + nl) * D_LEN;
        const int kh = hi * 16;
        #pragma unroll
        for (int r = 0; r < 8; ++r) {
            bk0[2*r+0] = (_Float16)kp[kh + 2*r + 0];
            bk0[2*r+1] = (_Float16)kp[kh + 2*r + 1];
            bk1[2*r+0] = (_Float16)kp[32 + kh + 2*r + 0];
            bk1[2*r+1] = (_Float16)kp[32 + kh + 2*r + 1];
        }
    }

    float m_col = -__builtin_inff();
    float l_col = 0.0f;

    const int qrow = tid >> 4;          // 0..15  (coop loader row)
    const int qc4  = (tid & 15) * 4;    // 0..60  (coop loader column chunk)

    for (int q0 = 0; q0 < S_LEN; q0 += 16) {
        __syncthreads();  // previous iteration's readers done with sQ
        {   // cooperative: stage scaled Q tile (16x64) as f16, row-major
            const float4 f = *(const float4*)(Qh + (size_t)(q0 + qrow) * D_LEN + qc4);
            _Float16* d = &sQ[qrow * KPAD + qc4];
            d[0] = (_Float16)(f.x * inv_sf);
            d[1] = (_Float16)(f.y * inv_sf);
            d[2] = (_Float16)(f.z * inv_sf);
            d[3] = (_Float16)(f.w * inv_sf);
            if (q0 + 16 < S_LEN)
                __builtin_prefetch(Qh + (size_t)(q0 + 16 + qrow) * D_LEN + qc4, 0, 1);
        }
        __syncthreads();

        // A operand: row = lane&15; element pairs are contiguous d-runs in LDS.
        const _Float16* qr = &sQ[nl * KPAD];
        const v16h aq0 = cat8(ld8(qr +      hi * 16), ld8(qr +  8 + hi * 16));
        const v16h aq1 = cat8(ld8(qr + 32 + hi * 16), ld8(qr + 40 + hi * 16));

        v8f c = {};
        c = __builtin_amdgcn_wmma_f32_16x16x32_f16(false, aq0, false, bk0,
                                                   (short)0, c, false, false);
        c = __builtin_amdgcn_wmma_f32_16x16x32_f16(false, aq1, false, bk1,
                                                   (short)0, c, false, false);

        // C layout: lane&15 = column n, VGPR r = row (hi*8 + r).
        float tmax = c[0];
        #pragma unroll
        for (int r = 1; r < 8; ++r) tmax = fmaxf(tmax, c[r]);
        tmax = fmaxf(tmax, __shfl_xor(tmax, 16, 32));
        const float new_m = fmaxf(m_col, tmax);
        float s = 0.0f;
        #pragma unroll
        for (int r = 0; r < 8; ++r) s += __expf(c[r] - new_m);
        s += __shfl_xor(s, 16, 32);
        l_col = l_col * __expf(m_col - new_m) + s;
        m_col = new_m;
    }
    if (lane < 16) {
        m_ws[(size_t)head * S_LEN + kb0 + nl] = m_col;
        l_ws[(size_t)head * S_LEN + kb0 + nl] = l_col;
    }
}

// -------- Pass 2: out[q,:] = sum_k exp(s[q,k]-m[k])/l[k] * V[k,:] ----------
__global__ __launch_bounds__(256) void
attn_output_kernel(const float* __restrict__ Q, const float* __restrict__ K,
                   const float* __restrict__ V, const float* __restrict__ sf_p,
                   const float* __restrict__ m_ws, const float* __restrict__ l_ws,
                   float* __restrict__ Out)
{
    __shared__ _Float16 sK [32 * KPAD];        // K tile, row-major [k][d]
    __shared__ _Float16 sVt[64 * VPAD];        // V tile, transposed [d][k]
    __shared__ _Float16 sP [WAVES * 16 * VPAD];// per-wave P tile [q][k], padded

    const int tid  = threadIdx.x;
    const int lane = tid & 31;
    const int wave = tid >> 5;
    const int head = blockIdx.x / BLOCKS_PER_HEAD;
    const int qb0  = (blockIdx.x % BLOCKS_PER_HEAD) * (16 * WAVES) + wave * 16;

    const float inv_sf = 1.0f / sf_p[0];
    const size_t hb = (size_t)head * S_LEN * D_LEN;
    const float* __restrict__ Qh = Q + hb;
    const float* __restrict__ Kh = K + hb;
    const float* __restrict__ Vh = V + hb;
    const float* __restrict__ mh = m_ws + (size_t)head * S_LEN;
    const float* __restrict__ lh = l_ws + (size_t)head * S_LEN;
    _Float16* __restrict__ myP = sP + wave * (16 * VPAD);

    const int nl = lane & 15;
    const int hi = (lane >> 4) & 1;

    // A operand: this wave's 16 query rows (loaded once from global, scaled).
    v16h aq0, aq1;
    {
        const float* qp = Qh + (size_t)(qb0 + nl) * D_LEN;
        #pragma unroll
        for (int r = 0; r < 8; ++r) {
            const int kk = (r << 1) + (hi << 4);
            aq0[2*r+0] = (_Float16)(qp[kk + 0] * inv_sf);
            aq0[2*r+1] = (_Float16)(qp[kk + 1] * inv_sf);
            aq1[2*r+0] = (_Float16)(qp[32 + kk + 0] * inv_sf);
            aq1[2*r+1] = (_Float16)(qp[32 + kk + 1] * inv_sf);
        }
    }

    v8f o0 = {}, o1 = {}, o2 = {}, o3 = {};

    const int crow = tid >> 3;          // 0..31 (coop loader row)
    const int cc8  = (tid & 7) * 8;     // 0..56 (coop loader 8-wide d chunk)

    for (int kb = 0; kb < S_LEN; kb += 32) {
        __syncthreads();  // previous iteration's readers done with sK/sVt
        {   // cooperative: stage K (row-major) and V (transposed) tiles as f16
            const float* gk = Kh + (size_t)(kb + crow) * D_LEN + cc8;
            const float4 k0 = *(const float4*)(gk);
            const float4 k1 = *(const float4*)(gk + 4);
            _Float16* dk = &sK[crow * KPAD + cc8];
            dk[0] = (_Float16)k0.x; dk[1] = (_Float16)k0.y;
            dk[2] = (_Float16)k0.z; dk[3] = (_Float16)k0.w;
            dk[4] = (_Float16)k1.x; dk[5] = (_Float16)k1.y;
            dk[6] = (_Float16)k1.z; dk[7] = (_Float16)k1.w;

            const float* gv = Vh + (size_t)(kb + crow) * D_LEN + cc8;
            const float4 v0 = *(const float4*)(gv);
            const float4 v1 = *(const float4*)(gv + 4);
            sVt[(cc8 + 0) * VPAD + crow] = (_Float16)v0.x;
            sVt[(cc8 + 1) * VPAD + crow] = (_Float16)v0.y;
            sVt[(cc8 + 2) * VPAD + crow] = (_Float16)v0.z;
            sVt[(cc8 + 3) * VPAD + crow] = (_Float16)v0.w;
            sVt[(cc8 + 4) * VPAD + crow] = (_Float16)v1.x;
            sVt[(cc8 + 5) * VPAD + crow] = (_Float16)v1.y;
            sVt[(cc8 + 6) * VPAD + crow] = (_Float16)v1.z;
            sVt[(cc8 + 7) * VPAD + crow] = (_Float16)v1.w;

            if (kb + 32 < S_LEN) {
                __builtin_prefetch(gk + 32 * D_LEN, 0, 1);
                __builtin_prefetch(gv + 32 * D_LEN, 0, 1);
            }
        }
        __syncthreads();

        // ---- score tiles for key columns kb+nl and kb+16+nl ----
        // B operand chunks are contiguous d-runs of the key row in sK.
        const _Float16* kr0 = &sK[nl * KPAD + hi * 16];
        const _Float16* kr1 = &sK[(16 + nl) * KPAD + hi * 16];
        const v16h b0a = cat8(ld8(kr0),      ld8(kr0 + 8));
        const v16h b0b = cat8(ld8(kr0 + 32), ld8(kr0 + 40));
        const v16h b1a = cat8(ld8(kr1),      ld8(kr1 + 8));
        const v16h b1b = cat8(ld8(kr1 + 32), ld8(kr1 + 40));

        v8f c0 = {}, c1 = {};
        c0 = __builtin_amdgcn_wmma_f32_16x16x32_f16(false, aq0, false, b0a,
                                                    (short)0, c0, false, false);
        c0 = __builtin_amdgcn_wmma_f32_16x16x32_f16(false, aq1, false, b0b,
                                                    (short)0, c0, false, false);
        c1 = __builtin_amdgcn_wmma_f32_16x16x32_f16(false, aq0, false, b1a,
                                                    (short)0, c1, false, false);
        c1 = __builtin_amdgcn_wmma_f32_16x16x32_f16(false, aq1, false, b1b,
                                                    (short)0, c1, false, false);

        // ---- P = exp(S - m[k]) / l[k], staged to per-wave LDS tile ----
        {
            const float m0 = mh[kb + nl],      rl0 = 1.0f / lh[kb + nl];
            const float m1 = mh[kb + 16 + nl], rl1 = 1.0f / lh[kb + 16 + nl];
            #pragma unroll
            for (int r = 0; r < 8; ++r) {
                const int row = hi * 8 + r;
                myP[row * VPAD + nl]      = (_Float16)(__expf(c0[r] - m0) * rl0);
                myP[row * VPAD + 16 + nl] = (_Float16)(__expf(c1[r] - m1) * rl1);
            }
        }
        // intra-wave DS ordering makes the store->load below safe (per-wave tile)

        // ---- A operand for PV: P tile 16(q) x 32(k), contiguous k-runs ----
        const _Float16* pr = &myP[nl * VPAD];
        const v16h ap = cat8(ld8(pr + hi * 16), ld8(pr + 8 + hi * 16));

        // ---- 4 PV WMMAs: B operand = contiguous k-runs of transposed V ----
        #pragma unroll
        for (int t = 0; t < 4; ++t) {
            const _Float16* vv = &sVt[(t * 16 + nl) * VPAD + hi * 16];
            const v16h bv = cat8(ld8(vv), ld8(vv + 8));
            if      (t == 0) o0 = __builtin_amdgcn_wmma_f32_16x16x32_f16(false, ap, false, bv, (short)0, o0, false, false);
            else if (t == 1) o1 = __builtin_amdgcn_wmma_f32_16x16x32_f16(false, ap, false, bv, (short)0, o1, false, false);
            else if (t == 2) o2 = __builtin_amdgcn_wmma_f32_16x16x32_f16(false, ap, false, bv, (short)0, o2, false, false);
            else             o3 = __builtin_amdgcn_wmma_f32_16x16x32_f16(false, ap, false, bv, (short)0, o3, false, false);
        }
    }

    // ---- write 16x64 output tile (C layout -> row-major global) ----
    float* op = Out + hb;
    #pragma unroll
    for (int r = 0; r < 8; ++r) {
        const int row = qb0 + hi * 8 + r;
        op[(size_t)row * D_LEN +  0 + nl] = o0[r];
        op[(size_t)row * D_LEN + 16 + nl] = o1[r];
        op[(size_t)row * D_LEN + 32 + nl] = o2[r];
        op[(size_t)row * D_LEN + 48 + nl] = o3[r];
    }
}

extern "C" void kernel_launch(void* const* d_in, const int* in_sizes, int n_in,
                              void* d_out, int out_size, void* d_ws, size_t ws_size,
                              hipStream_t stream) {
    (void)in_sizes; (void)n_in; (void)out_size; (void)ws_size;
    const float* Q  = (const float*)d_in[0];
    const float* K  = (const float*)d_in[1];
    const float* V  = (const float*)d_in[2];
    const float* sf = (const float*)d_in[3];
    float* out  = (float*)d_out;
    float* m_ws = (float*)d_ws;                       // NHEAD*S_LEN floats
    float* l_ws = m_ws + (size_t)NHEAD * S_LEN;       // NHEAD*S_LEN floats

    const dim3 block(32 * WAVES);
    const dim3 grid(NHEAD * BLOCKS_PER_HEAD);         // 512 workgroups

    attn_colstats_kernel<<<grid, block, 0, stream>>>(Q, K, sf, m_ws, l_ws);
    attn_output_kernel <<<grid, block, 0, stream>>>(Q, K, V, sf, m_ws, l_ws, out);
}
